// LPropDecoderLayer_56951266345699
// MI455X (gfx1250) — compile-verified, matrix operand order checked by reading, not run on hardware
//
#include <hip/hip_runtime.h>
#include <hip/hip_bf16.h>

typedef __attribute__((ext_vector_type(16))) __bf16 v16bf;
typedef __attribute__((ext_vector_type(8)))  float  v8f;

#define WMMA_BF16(A,B,C) __builtin_amdgcn_wmma_f32_16x16x32_bf16(false,(A),false,(B),(short)0,(C),false,false)

// Load a WMMA A/B operand (16 bf16 elements) for one lane.
// Row-major matrix; rowbase points at the row start.
// ISA 16-bit layout: lanes 0-15 (hf=0): K = {k0..k0+7, k0+16..k0+23}
//                    lanes 16-31 (hf=1): K = {k0+8..k0+15, k0+24..k0+31}
__device__ __forceinline__ v16bf ld_op16(const __hip_bfloat16* rowbase, int k0, int hf) {
  const __hip_bfloat16* p = rowbase + k0 + hf * 8;
  union { uint4 u[2]; v16bf v; } w;
  w.u[0] = *(const uint4*)(p);
  w.u[1] = *(const uint4*)(p + 16);
  return w.v;
}

__device__ __forceinline__ float redmax16(float x) {
  #pragma unroll
  for (int m = 1; m < 16; m <<= 1) x = fmaxf(x, __shfl_xor(x, m, 32));
  return x;
}
__device__ __forceinline__ float redsum16(float x) {
  #pragma unroll
  for (int m = 1; m < 16; m <<= 1) x += __shfl_xor(x, m, 32);
  return x;
}

// ---------------------------------------------------------------------------
// (F,S,C,HW) f32 -> (S, F*HW, C) bf16 (position-major rows = WMMA A rows)
__global__ void pack_posmajor_bf16(const float* __restrict__ in, __hip_bfloat16* __restrict__ out,
                                   int F, int S, int C, int HW) {
  size_t i = (size_t)blockIdx.x * blockDim.x + threadIdx.x;
  size_t total = (size_t)F * S * C * HW;
  if (i >= total) return;
  int ch = (int)(i % C);
  size_t r = i / C;
  int l = (int)(r % ((size_t)F * HW));
  int s = (int)(r / ((size_t)F * HW));
  int f = l / HW, p = l % HW;
  out[i] = __float2bfloat16(in[((((size_t)f * S + s) * C + ch) * HW) + p]);
}

__global__ void cvt_f32_bf16(const float* __restrict__ in, __hip_bfloat16* __restrict__ out, int n) {
  int i = blockIdx.x * blockDim.x + threadIdx.x;
  if (i < n) out[i] = __float2bfloat16(in[i]);
}

// v2T[s][e][l] = sum_j ca_wv_w[e][j] * pos_enc[f][s][j][p] + b[e]   (l = f*HW+p)
__global__ void v_proj_T(const float* __restrict__ pos, const float* __restrict__ wv,
                         const float* __restrict__ bias, __hip_bfloat16* __restrict__ out,
                         int F, int S, int HW) {
  int L = F * HW;
  size_t i = (size_t)blockIdx.x * blockDim.x + threadIdx.x;
  size_t total = (size_t)S * 16 * L;
  if (i >= total) return;
  int l = (int)(i % L);
  int r = (int)(i / L);
  int e = r % 16, s = r / 16;
  int f = l / HW, p = l % HW;
  float acc = bias[e];
  #pragma unroll
  for (int j = 0; j < 16; ++j)
    acc += wv[e * 16 + j] * pos[((((size_t)f * S + s) * 16 + j) * HW) + p];
  out[i] = __float2bfloat16(acc);
}

// ---------------------------------------------------------------------------
// C[M, bn0..bn0+16*NT] = A[M,K] * B[N,K]^T + bias. One wave computes a 16x(16*NT)
// strip: each A chunk loaded once, reused across NT column tiles (NT independent
// accumulator chains -> good WMMA back-to-back issue).
template <int NT>
__global__ void gemm_abT_nt(const __hip_bfloat16* __restrict__ A, const __hip_bfloat16* __restrict__ B,
                            const float* __restrict__ bias, float* __restrict__ C,
                            int M, int N, int K, long sA, long sC) {
  int lane = threadIdx.x, hf = lane >> 4, col = lane & 15;
  int bm = blockIdx.x * 16, bn0 = blockIdx.y * 16 * NT;
  const __hip_bfloat16* Ab = A + (size_t)blockIdx.z * sA;
  float* Cb = C + (size_t)blockIdx.z * sC;
  const __hip_bfloat16* arow = Ab + (size_t)(bm + col) * K;
  v8f acc[NT];
  #pragma unroll
  for (int n = 0; n < NT; ++n) acc[n] = (v8f){};
  for (int k0 = 0; k0 < K; k0 += 32) {
    v16bf a = ld_op16(arow, k0, hf);
    #pragma unroll
    for (int n = 0; n < NT; ++n) {
      v16bf b = ld_op16(B + (size_t)(bn0 + n * 16 + col) * K, k0, hf);
      acc[n] = WMMA_BF16(a, b, acc[n]);
    }
  }
  #pragma unroll
  for (int n = 0; n < NT; ++n) {
    int bn = bn0 + n * 16;
    float bv = bias ? bias[bn + col] : 0.0f;
    #pragma unroll
    for (int v = 0; v < 8; ++v)
      Cb[(size_t)(bm + v + 8 * hf) * N + bn + col] = acc[n][v] + bv;
  }
}

// Same, but stores C transposed as bf16: CT[N,M] (attention V^T operand).
template <int NT>
__global__ void gemm_abT_nt_storeT(const __hip_bfloat16* __restrict__ A, const __hip_bfloat16* __restrict__ B,
                                   const float* __restrict__ bias, __hip_bfloat16* __restrict__ CT,
                                   int M, int N, int K, long sA, long sC) {
  int lane = threadIdx.x, hf = lane >> 4, col = lane & 15;
  int bm = blockIdx.x * 16, bn0 = blockIdx.y * 16 * NT;
  const __hip_bfloat16* Ab = A + (size_t)blockIdx.z * sA;
  __hip_bfloat16* Cb = CT + (size_t)blockIdx.z * sC;
  const __hip_bfloat16* arow = Ab + (size_t)(bm + col) * K;
  v8f acc[NT];
  #pragma unroll
  for (int n = 0; n < NT; ++n) acc[n] = (v8f){};
  for (int k0 = 0; k0 < K; k0 += 32) {
    v16bf a = ld_op16(arow, k0, hf);
    #pragma unroll
    for (int n = 0; n < NT; ++n) {
      v16bf b = ld_op16(B + (size_t)(bn0 + n * 16 + col) * K, k0, hf);
      acc[n] = WMMA_BF16(a, b, acc[n]);
    }
  }
  #pragma unroll
  for (int n = 0; n < NT; ++n) {
    int bn = bn0 + n * 16;
    float bv = bias ? bias[bn + col] : 0.0f;
    #pragma unroll
    for (int v = 0; v < 8; ++v)
      Cb[(size_t)(bn + col) * M + bm + v + 8 * hf] = __float2bfloat16(acc[n][v] + bv);
  }
}

// Row-wise l2 normalize rows of 128, f32 in -> bf16 out. One 128-thread block per row.
__global__ void l2norm128(const float* __restrict__ X, __hip_bfloat16* __restrict__ Y) {
  __shared__ float sh[128];
  int r = blockIdx.x, t = threadIdx.x;
  float x = X[(size_t)r * 128 + t];
  sh[t] = x * x;
  __syncthreads();
  for (int st = 64; st > 0; st >>= 1) {
    if (t < st) sh[t] += sh[t + st];
    __syncthreads();
  }
  float sc = 1.0f / fmaxf(sqrtf(sh[0]), 1e-12f);
  Y[(size_t)r * 128 + t] = __float2bfloat16(x * sc);
}

// InstanceNorm over 1024 positions per (s,ch). U layout (S,HW,C) f32 -> bf16 same layout.
__global__ void inorm1024(const float* __restrict__ U, __hip_bfloat16* __restrict__ Y, int C) {
  int s = blockIdx.x / C, ch = blockIdx.x % C;
  int t = threadIdx.x;  // 256 threads
  __shared__ float s1[256], s2[256];
  float vals[4], a = 0.f, b = 0.f;
  #pragma unroll
  for (int j = 0; j < 4; ++j) {
    int p = t + j * 256;
    float u = U[((size_t)s * 1024 + p) * C + ch];
    vals[j] = u; a += u; b += u * u;
  }
  s1[t] = a; s2[t] = b; __syncthreads();
  for (int st = 128; st > 0; st >>= 1) {
    if (t < st) { s1[t] += s1[t + st]; s2[t] += s2[t + st]; }
    __syncthreads();
  }
  float mean = s1[0] * (1.0f / 1024.0f);
  float var  = s2[0] * (1.0f / 1024.0f) - mean * mean;
  float rstd = rsqrtf(var + 1e-5f);
  #pragma unroll
  for (int j = 0; j < 4; ++j) {
    int p = t + j * 256;
    Y[((size_t)s * 1024 + p) * C + ch] = __float2bfloat16((vals[j] - mean) * rstd);
  }
}

// ---------------------------------------------------------------------------
// Split-K flash attention core for one wave: processes keys [kb0, kb0+klen),
// maintains online softmax over NT output tiles in registers.
// Self-attn: 4 waves/block, each owns 256 of 1024 keys, dv=256 (NT=16).
// Partials merged via LDS; residual added in the combine step.
__global__ void __launch_bounds__(128) sa_flash_wg(
    const __hip_bfloat16* __restrict__ QN, const __hip_bfloat16* __restrict__ KN,
    const __hip_bfloat16* __restrict__ VT, const float* __restrict__ resid,
    float* __restrict__ U) {
  const int KD = 128, LK = 1024, HW = 1024, CH = 256, NT = 16, NW = 4;
  int tid = threadIdx.x;
  int wv = tid >> 5, lane = tid & 31, hf = lane >> 4, col = lane & 15;
  int qb = blockIdx.x * 16, s = blockIdx.y;

  __shared__ __align__(16) __hip_bfloat16 pbuf[NW][16 * 32];
  __shared__ float om[NW][16], os[NW][16];
  __shared__ float oacc[NW][16][CH];

  const __hip_bfloat16* Qs = QN + (size_t)s * HW * KD;
  const __hip_bfloat16* Ks = KN + (size_t)s * LK * KD;
  const __hip_bfloat16* Vs = VT + (size_t)s * CH * LK;

  v16bf qa[4];
  #pragma unroll
  for (int c4 = 0; c4 < 4; ++c4)
    qa[c4] = ld_op16(Qs + (size_t)(qb + col) * KD, c4 * 32, hf);

  v8f o[NT];
  #pragma unroll
  for (int n = 0; n < NT; ++n) o[n] = (v8f){};
  float rm[8], rs[8];
  #pragma unroll
  for (int v = 0; v < 8; ++v) { rm[v] = -1e30f; rs[v] = 0.0f; }

  int kb0 = wv * (LK / NW), kb1 = kb0 + (LK / NW);
  for (int kb = kb0; kb < kb1; kb += 32) {
    v8f sc0 = {}, sc1 = {};
    #pragma unroll
    for (int c4 = 0; c4 < 4; ++c4)
      sc0 = WMMA_BF16(qa[c4], ld_op16(Ks + (size_t)(kb + col) * KD, c4 * 32, hf), sc0);
    #pragma unroll
    for (int c4 = 0; c4 < 4; ++c4)
      sc1 = WMMA_BF16(qa[c4], ld_op16(Ks + (size_t)(kb + 16 + col) * KD, c4 * 32, hf), sc1);

    #pragma unroll
    for (int v = 0; v < 8; ++v) {
      float a0 = sc0[v] * 30.0f, a1 = sc1[v] * 30.0f;  // 1/tau = 30
      float mx = redmax16(fmaxf(a0, a1));
      float mn = fmaxf(rm[v], mx);
      float alpha = __expf(rm[v] - mn);
      rm[v] = mn;
      float p0 = __expf(a0 - mn);
      float p1 = __expf(a1 - mn);
      rs[v] = rs[v] * alpha + redsum16(p0 + p1);
      #pragma unroll
      for (int n = 0; n < NT; ++n) o[n][v] *= alpha;
      pbuf[wv][(v + 8 * hf) * 32 + col]      = __float2bfloat16(p0);
      pbuf[wv][(v + 8 * hf) * 32 + 16 + col] = __float2bfloat16(p1);
    }
    asm volatile("s_wait_dscnt 0" ::: "memory");
    v16bf pa = ld_op16(&pbuf[wv][col * 32], 0, hf);
    #pragma unroll
    for (int n = 0; n < NT; ++n)
      o[n] = WMMA_BF16(pa, ld_op16(Vs + (size_t)(n * 16 + col) * LK, kb, hf), o[n]);
    asm volatile("s_wait_dscnt 0" ::: "memory");
  }

  // Publish this wave's partial (unnormalized O + row stats) to LDS.
  if (col == 0) {
    #pragma unroll
    for (int v = 0; v < 8; ++v) { om[wv][v + 8 * hf] = rm[v]; os[wv][v + 8 * hf] = rs[v]; }
  }
  #pragma unroll
  for (int n = 0; n < NT; ++n)
    #pragma unroll
    for (int v = 0; v < 8; ++v)
      oacc[wv][v + 8 * hf][n * 16 + col] = o[n][v];
  __syncthreads();

  // Combine partials, normalize, add residual, write U (S,HW,C).
  for (int idx = tid; idx < 16 * CH; idx += 128) {
    int r = idx >> 8, ch = idx & 255;
    float M = -1e30f;
    #pragma unroll
    for (int p = 0; p < NW; ++p) M = fmaxf(M, om[p][r]);
    float Ssum = 0.f, val = 0.f;
    #pragma unroll
    for (int p = 0; p < NW; ++p) {
      float wgt = __expf(om[p][r] - M);
      Ssum += os[p][r] * wgt;
      val  += oacc[p][r][ch] * wgt;
    }
    int m = qb + r;
    float rr = resid[((size_t)s * CH + ch) * HW + m];
    U[((size_t)s * HW + m) * CH + ch] = val / Ssum + rr;
  }
}

// Cross-attn: 8 waves/block, each owns 1024 of 8192 keys, dv=16 (one tile).
// Writes (1,S,16,32,32) output directly.
__global__ void __launch_bounds__(256) ca_flash_wg(
    const __hip_bfloat16* __restrict__ QN, const __hip_bfloat16* __restrict__ KN,
    const __hip_bfloat16* __restrict__ VT, float* __restrict__ out) {
  const int KD = 128, LK = 8192, HW = 1024, CE = 16, NW = 8;
  int tid = threadIdx.x;
  int wv = tid >> 5, lane = tid & 31, hf = lane >> 4, col = lane & 15;
  int qb = blockIdx.x * 16, s = blockIdx.y;

  __shared__ __align__(16) __hip_bfloat16 pbuf[NW][16 * 32];
  __shared__ float om[NW][16], os[NW][16];
  __shared__ float oacc[NW][16][CE];

  const __hip_bfloat16* Qs = QN + (size_t)s * HW * KD;
  const __hip_bfloat16* Ks = KN + (size_t)s * LK * KD;
  const __hip_bfloat16* Vs = VT + (size_t)s * CE * LK;

  v16bf qa[4];
  #pragma unroll
  for (int c4 = 0; c4 < 4; ++c4)
    qa[c4] = ld_op16(Qs + (size_t)(qb + col) * KD, c4 * 32, hf);

  v8f o = {};
  float rm[8], rs[8];
  #pragma unroll
  for (int v = 0; v < 8; ++v) { rm[v] = -1e30f; rs[v] = 0.0f; }

  int kb0 = wv * (LK / NW), kb1 = kb0 + (LK / NW);
  for (int kb = kb0; kb < kb1; kb += 32) {
    v8f sc0 = {}, sc1 = {};
    #pragma unroll
    for (int c4 = 0; c4 < 4; ++c4)
      sc0 = WMMA_BF16(qa[c4], ld_op16(Ks + (size_t)(kb + col) * KD, c4 * 32, hf), sc0);
    #pragma unroll
    for (int c4 = 0; c4 < 4; ++c4)
      sc1 = WMMA_BF16(qa[c4], ld_op16(Ks + (size_t)(kb + 16 + col) * KD, c4 * 32, hf), sc1);

    #pragma unroll
    for (int v = 0; v < 8; ++v) {
      float a0 = sc0[v] * 30.0f, a1 = sc1[v] * 30.0f;
      float mx = redmax16(fmaxf(a0, a1));
      float mn = fmaxf(rm[v], mx);
      float alpha = __expf(rm[v] - mn);
      rm[v] = mn;
      float p0 = __expf(a0 - mn);
      float p1 = __expf(a1 - mn);
      rs[v] = rs[v] * alpha + redsum16(p0 + p1);
      o[v] *= alpha;
      pbuf[wv][(v + 8 * hf) * 32 + col]      = __float2bfloat16(p0);
      pbuf[wv][(v + 8 * hf) * 32 + 16 + col] = __float2bfloat16(p1);
    }
    asm volatile("s_wait_dscnt 0" ::: "memory");
    v16bf pa = ld_op16(&pbuf[wv][col * 32], 0, hf);
    o = WMMA_BF16(pa, ld_op16(Vs + (size_t)col * LK, kb, hf), o);
    asm volatile("s_wait_dscnt 0" ::: "memory");
  }

  if (col == 0) {
    #pragma unroll
    for (int v = 0; v < 8; ++v) { om[wv][v + 8 * hf] = rm[v]; os[wv][v + 8 * hf] = rs[v]; }
  }
  #pragma unroll
  for (int v = 0; v < 8; ++v)
    oacc[wv][v + 8 * hf][col] = o[v];
  __syncthreads();

  // 256 threads == 16 rows x 16 channels: combine + normalize + store.
  {
    int r = tid >> 4, e = tid & 15;
    float M = -1e30f;
    #pragma unroll
    for (int p = 0; p < NW; ++p) M = fmaxf(M, om[p][r]);
    float Ssum = 0.f, val = 0.f;
    #pragma unroll
    for (int p = 0; p < NW; ++p) {
      float wgt = __expf(om[p][r] - M);
      Ssum += os[p][r] * wgt;
      val  += oacc[p][r][e] * wgt;
    }
    out[((size_t)s * CE + e) * HW + qb + r] = val / Ssum;
  }
}

// ---------------------------------------------------------------------------
extern "C" void kernel_launch(void* const* d_in, const int* in_sizes, int n_in,
                              void* d_out, int out_size, void* d_ws, size_t ws_size,
                              hipStream_t stream) {
  const float* tgt     = (const float*)d_in[0];
  const float* memory_ = (const float*)d_in[1];
  const float* pos_enc = (const float*)d_in[2];
  const float* sa_wk_w = (const float*)d_in[3];
  const float* sa_wk_b = (const float*)d_in[4];
  const float* sa_wv_w = (const float*)d_in[5];
  const float* sa_wv_b = (const float*)d_in[6];
  const float* ca_wk_w = (const float*)d_in[7];
  const float* ca_wk_b = (const float*)d_in[8];
  const float* ca_wv_w = (const float*)d_in[9];
  const float* ca_wv_b = (const float*)d_in[10];
  (void)in_sizes; (void)n_in; (void)out_size; (void)ws_size;

  const int F = 8, S = 4, C = 256, HW = 1024, KD = 128, CE = 16, L = F * HW;

  char* wptr = (char*)d_ws;
  auto alloc = [&](size_t bytes) -> void* {
    void* p = (void*)wptr;
    wptr += (bytes + 255) & ~(size_t)255;
    return p;
  };

  __hip_bfloat16* tbf  = (__hip_bfloat16*)alloc((size_t)S * HW * C * 2);
  __hip_bfloat16* mbf  = (__hip_bfloat16*)alloc((size_t)S * L * C * 2);
  __hip_bfloat16* wkSA = (__hip_bfloat16*)alloc((size_t)KD * C * 2);
  __hip_bfloat16* wvSA = (__hip_bfloat16*)alloc((size_t)C * C * 2);
  __hip_bfloat16* wkCA = (__hip_bfloat16*)alloc((size_t)KD * C * 2);
  float*          qraw = (float*)alloc((size_t)S * HW * KD * 4);
  __hip_bfloat16* qnSA = (__hip_bfloat16*)alloc((size_t)S * HW * KD * 2);
  __hip_bfloat16* vTSA = (__hip_bfloat16*)alloc((size_t)S * C * HW * 2);
  float*          U    = (float*)alloc((size_t)S * HW * C * 4);
  __hip_bfloat16* t2bf = (__hip_bfloat16*)alloc((size_t)S * HW * C * 2);
  float*          q2r  = (float*)alloc((size_t)S * HW * KD * 4);
  __hip_bfloat16* q2bf = (__hip_bfloat16*)alloc((size_t)S * HW * KD * 2);
  float*          k2r  = (float*)alloc((size_t)S * L * KD * 4);
  __hip_bfloat16* k2bf = (__hip_bfloat16*)alloc((size_t)S * L * KD * 2);
  __hip_bfloat16* v2T  = (__hip_bfloat16*)alloc((size_t)S * CE * L * 2);

  // Pack inputs to position-major bf16; convert weights; project pos_enc values (stored V^T).
  { size_t tot = (size_t)1 * S * C * HW;
    pack_posmajor_bf16<<<(unsigned)((tot + 255) / 256), 256, 0, stream>>>(tgt, tbf, 1, S, C, HW); }
  { size_t tot = (size_t)F * S * C * HW;
    pack_posmajor_bf16<<<(unsigned)((tot + 255) / 256), 256, 0, stream>>>(memory_, mbf, F, S, C, HW); }
  cvt_f32_bf16<<<(KD * C + 255) / 256, 256, 0, stream>>>(sa_wk_w, wkSA, KD * C);
  cvt_f32_bf16<<<(C * C + 255) / 256, 256, 0, stream>>>(sa_wv_w, wvSA, C * C);
  cvt_f32_bf16<<<(KD * C + 255) / 256, 256, 0, stream>>>(ca_wk_w, wkCA, KD * C);
  { size_t tot = (size_t)S * CE * L;
    v_proj_T<<<(unsigned)((tot + 255) / 256), 256, 0, stream>>>(pos_enc, ca_wv_w, ca_wv_b, v2T, F, S, HW); }

  // Self-attention projections (WMMA) + l2norm.
  gemm_abT_nt<8><<<dim3(HW / 16, KD / 128, S), 32, 0, stream>>>(tbf, wkSA, sa_wk_b, qraw, HW, KD, C,
                                                                (long)HW * C, (long)HW * KD);
  l2norm128<<<S * HW, 128, 0, stream>>>(qraw, qnSA);
  gemm_abT_nt_storeT<8><<<dim3(HW / 16, C / 128, S), 32, 0, stream>>>(tbf, wvSA, sa_wv_b, vTSA, HW, C, C,
                                                                      (long)HW * C, (long)C * HW);

  // Self-attention (split-K flash, 4 waves/block) + residual in combine.
  sa_flash_wg<<<dim3(HW / 16, S), 128, 0, stream>>>(qnSA, qnSA, vTSA, tgt, U);

  // Instance norm.
  inorm1024<<<S * C, 256, 0, stream>>>(U, t2bf, C);

  // Cross-attention projections (WMMA) + l2norm.
  gemm_abT_nt<8><<<dim3(HW / 16, KD / 128, S), 32, 0, stream>>>(t2bf, wkCA, ca_wk_b, q2r, HW, KD, C,
                                                                (long)HW * C, (long)HW * KD);
  l2norm128<<<S * HW, 128, 0, stream>>>(q2r, q2bf);
  gemm_abT_nt<8><<<dim3(L / 16, KD / 128, S), 32, 0, stream>>>(mbf, wkCA, ca_wk_b, k2r, L, KD, C,
                                                               (long)L * C, (long)L * KD);
  l2norm128<<<S * L, 128, 0, stream>>>(k2r, k2bf);

  // Cross-attention (split-K flash, 8 waves/block) -> output (1,S,16,32,32).
  ca_flash_wg<<<dim3(HW / 16, S), 256, 0, stream>>>(q2bf, k2bf, v2T, (float*)d_out);
}
